// SampledMixtureLinear_73237782331624
// MI455X (gfx1250) — matrix operand order repaired and checked by reading, not compiled, not run on hardware
//
#include <hip/hip_runtime.h>

// Problem constants (fixed by the reference): B=8192, I=1024, O=1024, K=4, R=5
#define NB 8192
#define NI 1024
#define NO 1024
#define NK 4
#define NR 5

typedef __attribute__((ext_vector_type(16))) _Float16 v16h;
typedef __attribute__((ext_vector_type(8)))  _Float16 v8h;
typedef __attribute__((ext_vector_type(4)))  _Float16 v4h;
typedef __attribute__((ext_vector_type(8)))  float    v8f;
typedef __attribute__((ext_vector_type(4)))  float    v4f;

// ---------------------------------------------------------------------------
// softmax over the 4 mixture logits (tiny, recomputed per thread; logits hit
// constant cache)
// ---------------------------------------------------------------------------
__device__ __forceinline__ void softmax4(const float* __restrict__ logits, float mix[4]) {
  float l0 = logits[0], l1 = logits[1], l2 = logits[2], l3 = logits[3];
  float m  = fmaxf(fmaxf(l0, l1), fmaxf(l2, l3));
  float e0 = expf(l0 - m), e1 = expf(l1 - m), e2 = expf(l2 - m), e3 = expf(l3 - m);
  float inv = 1.0f / (e0 + e1 + e2 + e3);
  mix[0] = e0 * inv; mix[1] = e1 * inv; mix[2] = e2 * inv; mix[3] = e3 * inv;
}

// ---------------------------------------------------------------------------
// Fold sampled mixture weights into one effective [O,I] matrix, stored f16.
//   w_eff = sum_k mix[k] * (mu + exp(0.5*lv) * mean_r eps)
// One thread per 4 consecutive elements (float4 loads, bandwidth-bound).
// ---------------------------------------------------------------------------
__global__ __launch_bounds__(256) void fold_weights_kernel(
    const float* __restrict__ mix_logits,
    const float* __restrict__ wmu,    // [K,O,I]
    const float* __restrict__ wlv,    // [K,O,I]
    const float* __restrict__ eps_w,  // [R,K,O,I]
    _Float16*    __restrict__ weff)   // [O,I] f16
{
  const int OI = NO * NI;
  const int p  = (blockIdx.x * blockDim.x + threadIdx.x) * 4;
  if (p >= OI) return;

  float mix[4];
  softmax4(mix_logits, mix);

  v4f acc = {};
#pragma unroll
  for (int k = 0; k < NK; ++k) {
    const size_t base = (size_t)k * OI + p;
    v4f mu = *(const v4f*)(wmu + base);
    v4f lv = *(const v4f*)(wlv + base);
    v4f es = {};
#pragma unroll
    for (int r = 0; r < NR; ++r)
      es += *(const v4f*)(eps_w + (size_t)(r * NK + k) * OI + p);
    es *= (1.0f / NR);
    v4f sig;
#pragma unroll
    for (int j = 0; j < 4; ++j) sig[j] = expf(0.5f * lv[j]);
    acc += mix[k] * (mu + sig * es);
  }

  v4h h;
#pragma unroll
  for (int j = 0; j < 4; ++j) h[j] = (_Float16)acc[j];
  *(v4h*)(weff + p) = h;
}

// ---------------------------------------------------------------------------
// Fold effective bias [O] (f32, exact).
// ---------------------------------------------------------------------------
__global__ __launch_bounds__(256) void fold_bias_kernel(
    const float* __restrict__ mix_logits,
    const float* __restrict__ bmu,    // [K,O]
    const float* __restrict__ blv,    // [K,O]
    const float* __restrict__ eps_b,  // [R,K,O]
    float*       __restrict__ beff)   // [O]
{
  const int o = blockIdx.x * blockDim.x + threadIdx.x;
  if (o >= NO) return;

  float mix[4];
  softmax4(mix_logits, mix);

  float acc = 0.0f;
#pragma unroll
  for (int k = 0; k < NK; ++k) {
    float es = 0.0f;
#pragma unroll
    for (int r = 0; r < NR; ++r)
      es += eps_b[(size_t)(r * NK + k) * NO + o];
    es *= (1.0f / NR);
    acc += mix[k] * (bmu[(size_t)k * NO + o] + expf(0.5f * blv[(size_t)k * NO + o]) * es);
  }
  beff[o] = acc;
}

// ---------------------------------------------------------------------------
// Convert activations x [B,I] f32 -> f16 once (re-read many times from L2).
// ---------------------------------------------------------------------------
__global__ __launch_bounds__(256) void cvt_x_kernel(
    const float* __restrict__ x, _Float16* __restrict__ xh)
{
  const int p = (blockIdx.x * blockDim.x + threadIdx.x) * 4;
  if (p >= NB * NI) return;
  v4f v = *(const v4f*)(x + p);
  v4h h;
#pragma unroll
  for (int j = 0; j < 4; ++j) h[j] = (_Float16)v[j];
  *(v4h*)(xh + p) = h;
}

// ---------------------------------------------------------------------------
// WMMA GEMM: out[B,O] = xh[B,I] * weff[O,I]^T + beff
//
// Block: 256 threads = 8 wave32 waves, arranged 2(M) x 4(N).
// Wave tile: 64(M) x 32(N)  -> 4x2 accumulators of v8f.
// Block tile: 128 x 128; grid = (O/128, B/128) = (8, 64).
// K loop: steps of 32, v_wmma_f32_16x16x32_f16, f32 accumulate.
//
// A and B both use contiguous k order (lane half h holds k = 16h..16h+15), not the
// ISA table order (CDNA5 ISA 7.12.2), so both load straight from row-major memory
// with b128 vector loads; no transpose instructions required.
// ---------------------------------------------------------------------------
__global__ __launch_bounds__(256) void gemm_wmma_kernel(
    const _Float16* __restrict__ xh,    // [B, I] row-major f16
    const _Float16* __restrict__ wh,    // [O, I] row-major f16
    const float*    __restrict__ beff,  // [O]
    float*          __restrict__ out)   // [B, O]
{
  const int lane = threadIdx.x & 31;
  const int wave = threadIdx.x >> 5;
  const int wm   = wave & 1;          // 0..1  (M direction)
  const int wn   = wave >> 1;         // 0..3  (N direction)
  const int m0   = blockIdx.y * 128 + wm * 64;
  const int n0   = blockIdx.x * 128 + wn * 32;
  const int lr   = lane & 15;         // row (A) / col (B,D) within 16-tile
  const int hi   = lane >> 4;         // half-wave selector

  v8f acc[4][2];
#pragma unroll
  for (int mt = 0; mt < 4; ++mt)
#pragma unroll
    for (int nt = 0; nt < 2; ++nt) {
      v8f z = {};
      acc[mt][nt] = z;
    }

  const int aoff = hi * 16;   // A: lanes 16-31 hold K+16..K+31
  const int boff = hi * 16;   // B: lanes 16-31 hold K+16..K+31

  for (int kk = 0; kk < NI; kk += 32) {
    // A fragments: 16x32 f16, per-lane two contiguous 16B runs
    v16h a[4];
#pragma unroll
    for (int mt = 0; mt < 4; ++mt) {
      const _Float16* ap = xh + (size_t)(m0 + mt * 16 + lr) * NI + kk + aoff;
      v8h alo = *(const v8h*)ap;          // elements 0..7
      v8h ahi = *(const v8h*)(ap + 8);    // elements 8..15
      a[mt] = __builtin_shufflevector(alo, ahi,
                                      0, 1, 2, 3, 4, 5, 6, 7,
                                      8, 9, 10, 11, 12, 13, 14, 15);
    }
    // B fragments: 32x16 f16, per-lane one contiguous 32B run (row of W)
    v16h b[2];
#pragma unroll
    for (int nt = 0; nt < 2; ++nt) {
      const _Float16* bp = wh + (size_t)(n0 + nt * 16 + lr) * NI + kk + boff;
      b[nt] = *(const v16h*)bp;
    }
    // 8 WMMAs per K step
#pragma unroll
    for (int mt = 0; mt < 4; ++mt)
#pragma unroll
      for (int nt = 0; nt < 2; ++nt)
        acc[mt][nt] = __builtin_amdgcn_wmma_f32_16x16x32_f16(
            false, a[mt], false, b[nt], (short)0, acc[mt][nt], false, false);
  }

  // Epilogue: D layout -> (row = m0 + mt*16 + hi*8 + r, col = n0 + nt*16 + lr)
#pragma unroll
  for (int nt = 0; nt < 2; ++nt) {
    const int col  = n0 + nt * 16 + lr;
    const float bv = beff[col];
#pragma unroll
    for (int mt = 0; mt < 4; ++mt) {
      const int rowbase = m0 + mt * 16 + hi * 8;
#pragma unroll
      for (int r = 0; r < 8; ++r)
        out[(size_t)(rowbase + r) * NO + col] = acc[mt][nt][r] + bv;
    }
  }
}

// ---------------------------------------------------------------------------
// Host launcher
// ---------------------------------------------------------------------------
extern "C" void kernel_launch(void* const* d_in, const int* in_sizes, int n_in,
                              void* d_out, int out_size, void* d_ws, size_t ws_size,
                              hipStream_t stream) {
  (void)in_sizes; (void)n_in; (void)out_size; (void)ws_size;

  const float* x          = (const float*)d_in[0];  // [B,I]
  const float* mix_logits = (const float*)d_in[1];  // [K]
  const float* wmu        = (const float*)d_in[2];  // [K,O,I]
  const float* bmu        = (const float*)d_in[3];  // [K,O]
  const float* wlv        = (const float*)d_in[4];  // [K,O,I]
  const float* blv        = (const float*)d_in[5];  // [K,O]
  const float* eps_w      = (const float*)d_in[6];  // [R,K,O,I]
  const float* eps_b      = (const float*)d_in[7];  // [R,K,O]
  float* out = (float*)d_out;                       // [B,O]

  // Workspace layout: w_eff f16 (2 MiB) | b_eff f32 (4 KiB) | x f16 (16 MiB)
  char* ws = (char*)d_ws;
  _Float16* weff = (_Float16*)ws;
  size_t off = (size_t)NO * NI * sizeof(_Float16);
  float* beff = (float*)(ws + off);
  off += (size_t)NO * sizeof(float);
  off = (off + 255) & ~(size_t)255;
  _Float16* xh = (_Float16*)(ws + off);

  fold_weights_kernel<<<(NO * NI / 4 + 255) / 256, 256, 0, stream>>>(
      mix_logits, wmu, wlv, eps_w, weff);
  fold_bias_kernel<<<(NO + 255) / 256, 256, 0, stream>>>(
      mix_logits, bmu, blv, eps_b, beff);
  cvt_x_kernel<<<(NB * NI / 4 + 255) / 256, 256, 0, stream>>>(x, xh);

  dim3 grid(NO / 128, NB / 128);   // (8, 64)
  gemm_wmma_kernel<<<grid, 256, 0, stream>>>(xh, weff, beff, out);
}
